// SwinDAttention_16844861735479
// MI455X (gfx1250) — compile-verified, matrix operand order checked by reading, not run on hardware
//
#include <hip/hip_runtime.h>
#include <hip/hip_bf16.h>

#define WSZ 7
#define P   49
#define C   256
#define NH  8
#define HC  32
#define NG  4
#define CGC 64
#define LN_EPS 1e-5f
#define ATT_SCALE 0.17677669529663687f   // 32^-0.5

typedef __bf16        v16bf __attribute__((ext_vector_type(16)));
typedef float         v8f   __attribute__((ext_vector_type(8)));
typedef unsigned int  v4u   __attribute__((ext_vector_type(4)));

// ---- WMMA fragment helpers (wave32 layouts per CDNA5 ISA 7.12.2) ----
__device__ __forceinline__ v16bf ld_frag(const __bf16* base, int row0, int k0,
                                         int stride, int lane) {
  const __bf16* p = base + (row0 + (lane & 15)) * stride + k0 + ((lane >> 4) << 3);
  union { v16bf v; v4u u[2]; } r;
  r.u[0] = *reinterpret_cast<const v4u*>(p);
  r.u[1] = *reinterpret_cast<const v4u*>(p + 16);
  return r.v;
}

__device__ __forceinline__ v8f wmma_bf16(v16bf a, v16bf b, v8f c) {
  return __builtin_amdgcn_wmma_f32_16x16x32_bf16(false, a, false, b, (short)0, c,
                                                 false, false);
}

// CDNA5 async global->LDS copy (ASYNCcnt-tracked, bypasses VGPR data path).
__device__ __forceinline__ void async_g2l_b128(unsigned lds_off, const void* gbase,
                                               int byte_off) {
  asm volatile("global_load_async_to_lds_b128 %0, %1, %2"
               :
               : "v"(lds_off), "v"(byte_off), "s"(gbase)
               : "memory");
}
__device__ __forceinline__ void async_g2l_b32(unsigned lds_off, const void* gbase,
                                              int byte_off) {
  asm volatile("global_load_async_to_lds_b32 %0, %1, %2"
               :
               : "v"(lds_off), "v"(byte_off), "s"(gbase)
               : "memory");
}
__device__ __forceinline__ void wait_async0() {
  asm volatile("s_wait_asynccnt 0x0" ::: "memory");
}

// GEMM: D[p][o] = sum_c A[p][c] * Wb[o][c] + bias[o]; K-outer/M-inner so each
// global weight fragment is loaded once and reused for all 4 M tiles.
__device__ __forceinline__ void gemm_pc(const __bf16* A, const __bf16* Wb,
                                        const float* bias, __bf16* D,
                                        int lane, int wave) {
  const int MT[4] = {0, 16, 32, 33};
  v8f acc[4][2];
  #pragma unroll
  for (int mt = 0; mt < 4; ++mt) { acc[mt][0] = 0.0f; acc[mt][1] = 0.0f; }
  #pragma unroll
  for (int ks = 0; ks < 8; ++ks) {
    v16bf b0 = ld_frag(Wb, wave * 32,      ks * 32, C, lane);
    v16bf b1 = ld_frag(Wb, wave * 32 + 16, ks * 32, C, lane);
    #pragma unroll
    for (int mt = 0; mt < 4; ++mt) {
      v16bf af = ld_frag(A, MT[mt], ks * 32, C, lane);
      acc[mt][0] = wmma_bf16(af, b0, acc[mt][0]);
      acc[mt][1] = wmma_bf16(af, b1, acc[mt][1]);
    }
  }
  #pragma unroll
  for (int mt = 0; mt < 4; ++mt) {
    union { v8f v; float f[8]; } u0, u1; u0.v = acc[mt][0]; u1.v = acc[mt][1];
    const int mb = MT[mt] + ((lane >> 4) << 3);
    const int n0 = wave * 32 + (lane & 15);
    const float bb0 = bias[n0], bb1 = bias[n0 + 16];
    #pragma unroll
    for (int r = 0; r < 8; ++r) {
      D[(mb + r) * C + n0]      = (__bf16)(u0.f[r] + bb0);
      D[(mb + r) * C + n0 + 16] = (__bf16)(u1.f[r] + bb1);
    }
  }
}

// Same GEMM, transposed store D[o][p] with stride 64 (for V).
__device__ __forceinline__ void gemm_pc_T(const __bf16* A, const __bf16* Wb,
                                          const float* bias, __bf16* D,
                                          int lane, int wave) {
  const int MT[4] = {0, 16, 32, 33};
  v8f acc[4][2];
  #pragma unroll
  for (int mt = 0; mt < 4; ++mt) { acc[mt][0] = 0.0f; acc[mt][1] = 0.0f; }
  #pragma unroll
  for (int ks = 0; ks < 8; ++ks) {
    v16bf b0 = ld_frag(Wb, wave * 32,      ks * 32, C, lane);
    v16bf b1 = ld_frag(Wb, wave * 32 + 16, ks * 32, C, lane);
    #pragma unroll
    for (int mt = 0; mt < 4; ++mt) {
      v16bf af = ld_frag(A, MT[mt], ks * 32, C, lane);
      acc[mt][0] = wmma_bf16(af, b0, acc[mt][0]);
      acc[mt][1] = wmma_bf16(af, b1, acc[mt][1]);
    }
  }
  #pragma unroll
  for (int mt = 0; mt < 4; ++mt) {
    union { v8f v; float f[8]; } u0, u1; u0.v = acc[mt][0]; u1.v = acc[mt][1];
    const int mb = MT[mt] + ((lane >> 4) << 3);
    const int n0 = wave * 32 + (lane & 15);
    const float bb0 = bias[n0], bb1 = bias[n0 + 16];
    #pragma unroll
    for (int r = 0; r < 8; ++r) {
      D[n0 * 64 + (mb + r)]        = (__bf16)(u0.f[r] + bb0);
      D[(n0 + 16) * 64 + (mb + r)] = (__bf16)(u1.f[r] + bb1);
    }
  }
}

__global__ __launch_bounds__(256) void convert_weights_bf16(
    const float* __restrict__ Wq, const float* __restrict__ Wk,
    const float* __restrict__ Wv, const float* __restrict__ Wo,
    __bf16* __restrict__ out) {
  int i = blockIdx.x * 256 + threadIdx.x;
  if (i < C * C) {
    out[i]             = (__bf16)Wq[i];
    out[C * C + i]     = (__bf16)Wk[i];
    out[2 * C * C + i] = (__bf16)Wv[i];
    out[3 * C * C + i] = (__bf16)Wo[i];
  }
}

__global__ __launch_bounds__(256) void swin_dattn_fused(
    const float* __restrict__ x1, const float* __restrict__ x2,
    const float* __restrict__ bq, const float* __restrict__ bk,
    const float* __restrict__ bv, const float* __restrict__ bo,
    const float* __restrict__ dww, const float* __restrict__ dwb,
    const float* __restrict__ lng, const float* __restrict__ lnb,
    const float* __restrict__ pww,
    const __bf16* __restrict__ Wqb, const __bf16* __restrict__ Wkb,
    const __bf16* __restrict__ Wvb, const __bf16* __restrict__ Wob,
    float* __restrict__ xout, float* __restrict__ attnout,
    int B1, int ratio) {
  // LDS: 25088 + 25088 + 32768 + 50176 + 1568 = ~132 KB (one block / WGP)
  __shared__ alignas(16) __bf16 sU0[P * C];              // x1 bf16 -> xs -> out
  __shared__ alignas(16) __bf16 sQ[P * C];               // q [p][c]
  __shared__ alignas(16) __bf16 sKV[C * 64];             // k [p][c] then v [c][p64]
  __shared__ alignas(16) unsigned char sUn[NG * P * CGC * 4]; // sT / sX2 / sAttn
  __shared__ float sPos[NG * P * 2];

  const int tid  = threadIdx.x;
  const int lane = tid & 31;
  const int wave = tid >> 5;
  const int b1   = blockIdx.x;
  const int MT[4] = {0, 16, 32, 33};

  // Warm L2 with the bf16 weights (emits global_prefetch_b8).
  __builtin_prefetch(Wqb + tid * 256, 0, 3);
  __builtin_prefetch(Wkb + tid * 256, 0, 3);
  __builtin_prefetch(Wvb + tid * 256, 0, 3);
  __builtin_prefetch(Wob + tid * 256, 0, 3);

  v8f oacc[4][2];
  #pragma unroll
  for (int i = 0; i < 4; ++i) { oacc[i][0] = 0.0f; oacc[i][1] = 0.0f; }

  for (int tt = 0; tt < ratio; ++tt) {
    const int win = b1 * ratio + tt;
    const int x1i = (win >= B1) ? (win - B1) : win;

    // ---- stage 1: x1 window -> sU0 (bf16 [p][c]) ----
    {
      const float* x1w = x1 + (size_t)x1i * P * C;
      for (int i = tid; i < P * C; i += 256) sU0[i] = (__bf16)x1w[i];
    }
    __syncthreads();

    // ---- stage 2: q = x1 @ Wq^T + bq ----
    gemm_pc(sU0, Wqb, bq, sQ, lane, wave);
    __syncthreads();

    // ---- stage 3: depthwise 5x5 conv on q (group-channel c = tid) ----
    {
      float* sT = (float*)sUn;                 // [g*49+p][64]
      const int ch = tid & 63;
      const int g  = tid >> 6;
      float wdw[25];
      #pragma unroll
      for (int k = 0; k < 25; ++k) wdw[k] = dww[ch * 25 + k];
      const float bdw = dwb[ch];
      for (int p = 0; p < P; ++p) {
        const int py = p / 7, px = p % 7;
        float acc = bdw;
        #pragma unroll
        for (int ky = 0; ky < 5; ++ky) {
          const int y = py + ky - 2;
          if (y < 0 || y > 6) continue;
          #pragma unroll
          for (int kx = 0; kx < 5; ++kx) {
            const int x = px + kx - 2;
            if (x < 0 || x > 6) continue;
            acc += wdw[ky * 5 + kx] * (float)sQ[(y * 7 + x) * C + tid];
          }
        }
        sT[(g * P + p) * CGC + ch] = acc;
      }
    }
    __syncthreads();

    // ---- stage 4: LayerNorm(64) + exact GELU + 1x1 -> offsets -> sPos ----
    if (tid < NG * P) {
      const float* row = (const float*)sUn + tid * CGC;
      float s = 0.0f, s2 = 0.0f;
      for (int ch = 0; ch < CGC; ++ch) { float v = row[ch]; s += v; s2 += v * v; }
      const float mu = s / CGC;
      const float var = s2 / CGC - mu * mu;
      const float rstd = rsqrtf(var + LN_EPS);
      float o0 = 0.0f, o1 = 0.0f;
      for (int ch = 0; ch < CGC; ++ch) {
        const float tn = (row[ch] - mu) * rstd * lng[ch] + lnb[ch];
        const float ge = 0.5f * tn * (1.0f + erff(tn * 0.70710678118654752f));
        o0 += pww[ch] * ge;
        o1 += pww[CGC + ch] * ge;
      }
      const int p = tid % P;
      const float refy = (((p / 7) + 0.5f) / 7.0f) * 2.0f - 1.0f;
      const float refx = (((p % 7) + 0.5f) / 7.0f) * 2.0f - 1.0f;
      sPos[tid * 2 + 0] = tanhf(o0) * (2.0f / 7.0f) + refy;
      sPos[tid * 2 + 1] = tanhf(o1) * (2.0f / 7.0f) + refx;
    }
    __syncthreads();

    // ---- stage 5: x2 window -> sUn (f32, 50176B) via async global->LDS DMA ----
    {
      const float* x2w = x2 + (size_t)win * P * C;
      #pragma unroll
      for (int it = 0; it < 12; ++it) {           // 12 * 256 lanes * 16B = 49152B
        const int off = (it * 256 + tid) * 16;
        async_g2l_b128((unsigned)(uintptr_t)(sUn + off), x2w, off);
      }
      {                                           // tail: 1024B
        const int off = 49152 + tid * 4;
        async_g2l_b32((unsigned)(uintptr_t)(sUn + off), x2w, off);
      }
      wait_async0();
    }
    __syncthreads();

    // ---- stage 6: bilinear grid-sample -> sU0 (xs bf16 [p][c]) ----
    if (tid < NG * P) {
      const float* sX2 = (const float*)sUn;
      const int g = tid & 3, p = tid >> 2;
      const float posy = sPos[(g * P + p) * 2 + 0];
      const float posx = sPos[(g * P + p) * 2 + 1];
      const float ixf = (posx + 1.0f) * 3.0f;
      const float iyf = (posy + 1.0f) * 3.0f;
      const float x0 = floorf(ixf), y0 = floorf(iyf);
      const float x1f = x0 + 1.0f, y1f = y0 + 1.0f;
      const float wgt[4] = {(x1f - ixf) * (y1f - iyf), (ixf - x0) * (y1f - iyf),
                            (x1f - ixf) * (iyf - y0), (ixf - x0) * (iyf - y0)};
      const float cxs[4] = {x0, x1f, x0, x1f};
      const float cys[4] = {y0, y0, y1f, y1f};
      int rowi[4]; float wv[4];
      #pragma unroll
      for (int i = 0; i < 4; ++i) {
        const bool valid = (cxs[i] >= 0.0f) && (cxs[i] <= 6.0f) &&
                           (cys[i] >= 0.0f) && (cys[i] <= 6.0f);
        const int xi = (int)fminf(fmaxf(cxs[i], 0.0f), 6.0f);
        const int yi = (int)fminf(fmaxf(cys[i], 0.0f), 6.0f);
        rowi[i] = (yi * 7 + xi) * C;
        wv[i] = valid ? wgt[i] : 0.0f;
      }
      const int cb = g * CGC;
      for (int cc = 0; cc < CGC; ++cc) {
        const int c = cb + cc;
        const float v = wv[0] * sX2[rowi[0] + c] + wv[1] * sX2[rowi[1] + c] +
                        wv[2] * sX2[rowi[2] + c] + wv[3] * sX2[rowi[3] + c];
        sU0[p * C + c] = (__bf16)v;
      }
    }
    __syncthreads();

    // ---- stage 7a: k = xs @ Wk^T + bk -> sKV [p][c] ----
    gemm_pc(sU0, Wkb, bk, sKV, lane, wave);
    __syncthreads();

    // ---- stage 8a: scores (wave = head) + softmax + global attn write ----
    {
      __bf16* sAh = (__bf16*)sUn + wave * (P * 64);
      #pragma unroll
      for (int mt = 0; mt < 4; ++mt) {
        v16bf af = ld_frag(sQ, MT[mt], wave * HC, C, lane);
        #pragma unroll
        for (int nt = 0; nt < 4; ++nt) {
          v16bf bf = ld_frag(sKV, MT[nt], wave * HC, C, lane);
          v8f d = 0.0f;
          d = wmma_bf16(af, bf, d);
          union { v8f v; float f[8]; } u; u.v = d;
          const int mb = MT[mt] + ((lane >> 4) << 3);
          const int n = MT[nt] + (lane & 15);
          #pragma unroll
          for (int r = 0; r < 8; ++r)
            sAh[(mb + r) * 64 + n] = (__bf16)(u.f[r] * ATT_SCALE);
        }
      }
      float* gattn = attnout +
          (((size_t)b1 * ratio + tt) * NH + wave) * (size_t)(P * P);
      for (int row = lane; row < P; row += 32) {
        __bf16* sr = sAh + row * 64;
        float mx = -1e30f;
        for (int k = 0; k < P; ++k) mx = fmaxf(mx, (float)sr[k]);
        float ssum = 0.0f;
        for (int k = 0; k < P; ++k) ssum += __expf((float)sr[k] - mx);
        const float inv = 1.0f / ssum;
        float* gr = gattn + row * P;
        for (int k = 0; k < P; ++k) {
          const float pv = __expf((float)sr[k] - mx) * inv;
          gr[k] = pv;
          sr[k] = (__bf16)pv;
        }
        #pragma unroll
        for (int k = P; k < 64; ++k) sr[k] = (__bf16)0.0f;  // K-pad for PV GEMM
      }
    }
    __syncthreads();

    // ---- stage 7b: v = xs @ Wv^T + bv -> sKV transposed [c][p64], pad zeroed ----
    gemm_pc_T(sU0, Wvb, bv, sKV, lane, wave);
    for (int i = tid; i < C * 15; i += 256)
      sKV[(i / 15) * 64 + P + (i % 15)] = (__bf16)0.0f;
    __syncthreads();

    // ---- stage 8b: out += attn @ v^T, accumulated in VGPRs across tt ----
    {
      const __bf16* sAh = (const __bf16*)sUn + wave * (P * 64);
      #pragma unroll
      for (int mt = 0; mt < 4; ++mt) {
        #pragma unroll
        for (int kk = 0; kk < 2; ++kk) {
          v16bf af = ld_frag(sAh, MT[mt], kk * 32, 64, lane);
          #pragma unroll
          for (int ct = 0; ct < 2; ++ct) {
            v16bf bf = ld_frag(sKV, wave * HC + ct * 16, kk * 32, 64, lane);
            oacc[mt][ct] = wmma_bf16(af, bf, oacc[mt][ct]);
          }
        }
      }
    }
    __syncthreads();
  }  // tt

  // ---- stage summed attention output into sU0 (bf16 [p][c]) ----
  #pragma unroll
  for (int mt = 0; mt < 4; ++mt) {
    #pragma unroll
    for (int ct = 0; ct < 2; ++ct) {
      union { v8f v; float f[8]; } u; u.v = oacc[mt][ct];
      const int mb = MT[mt] + ((lane >> 4) << 3);
      const int n = wave * 32 + ct * 16 + (lane & 15);
      #pragma unroll
      for (int r = 0; r < 8; ++r) sU0[(mb + r) * C + n] = (__bf16)u.f[r];
    }
  }
  __syncthreads();

  // ---- final: x_out = out @ Wo^T + bo -> global f32 ----
  {
    v8f acc[4][2];
    #pragma unroll
    for (int mt = 0; mt < 4; ++mt) { acc[mt][0] = 0.0f; acc[mt][1] = 0.0f; }
    #pragma unroll
    for (int ks = 0; ks < 8; ++ks) {
      v16bf b0 = ld_frag(Wob, wave * 32,      ks * 32, C, lane);
      v16bf b1f = ld_frag(Wob, wave * 32 + 16, ks * 32, C, lane);
      #pragma unroll
      for (int mt = 0; mt < 4; ++mt) {
        v16bf af = ld_frag(sU0, MT[mt], ks * 32, C, lane);
        acc[mt][0] = wmma_bf16(af, b0, acc[mt][0]);
        acc[mt][1] = wmma_bf16(af, b1f, acc[mt][1]);
      }
    }
    #pragma unroll
    for (int mt = 0; mt < 4; ++mt) {
      union { v8f v; float f[8]; } u0, u1; u0.v = acc[mt][0]; u1.v = acc[mt][1];
      const int mb = MT[mt] + ((lane >> 4) << 3);
      const int n0 = wave * 32 + (lane & 15);
      const float bb0 = bo[n0], bb1 = bo[n0 + 16];
      #pragma unroll
      for (int r = 0; r < 8; ++r) {
        const int m = mb + r;  // always <= 48 with overlap tiling
        xout[((size_t)b1 * P + m) * C + n0]      = u0.f[r] + bb0;
        xout[((size_t)b1 * P + m) * C + n0 + 16] = u1.f[r] + bb1;
      }
    }
  }
}

extern "C" void kernel_launch(void* const* d_in, const int* in_sizes, int n_in,
                              void* d_out, int out_size, void* d_ws, size_t ws_size,
                              hipStream_t stream) {
  const float* x1  = (const float*)d_in[0];
  const float* x2  = (const float*)d_in[1];
  const float* Wq  = (const float*)d_in[2];
  const float* bq  = (const float*)d_in[3];
  const float* Wk  = (const float*)d_in[4];
  const float* bk  = (const float*)d_in[5];
  const float* Wv  = (const float*)d_in[6];
  const float* bv  = (const float*)d_in[7];
  const float* Wo  = (const float*)d_in[8];
  const float* bo  = (const float*)d_in[9];
  const float* dww = (const float*)d_in[10];
  const float* dwb = (const float*)d_in[11];
  const float* lng = (const float*)d_in[12];
  const float* lnb = (const float*)d_in[13];
  const float* pww = (const float*)d_in[14];

  const int B1    = in_sizes[0] / (P * C);
  const int B     = in_sizes[1] / (P * C);
  const int ratio = B / B1;

  __bf16* wbf = (__bf16*)d_ws;  // 4 * 256*256 bf16 = 512 KB (L2-resident)
  convert_weights_bf16<<<(C * C + 255) / 256, 256, 0, stream>>>(Wq, Wk, Wv, Wo, wbf);

  float* xout    = (float*)d_out;
  float* attnout = xout + (size_t)B1 * P * C;
  swin_dattn_fused<<<B1, 256, 0, stream>>>(
      x1, x2, bq, bk, bv, bo, dww, dwb, lng, lnb, pww,
      wbf, wbf + C * C, wbf + 2 * C * C, wbf + 3 * C * C,
      xout, attnout, B1, ratio);
}
